// MultiHeadAttention_V2_42666205118533
// MI455X (gfx1250) — matrix operand order repaired and checked by reading, not compiled, not run on hardware
//
#include <hip/hip_runtime.h>

// MHA forward, B=2 T=2048 D=1024 H=16 HD=64.
// Pipeline: (1) QKV projection GEMM f32->f16 via v_wmma_f32_16x16x32_f16,
//           (2) flash-attention (online softmax, no mask per reference),
//               with async global->LDS staging of Q/K tiles (ASYNCcnt path),
//           (3) output projection + bias -> f32.
// Workspace layout (f16): Q | K | V | ctx, each B*T*D halves => 32 MB total.

#define B_  2
#define T_  2048
#define D_  1024
#define H_  16
#define HD_ 64
#define M_  (B_ * T_)   // 4096

typedef __attribute__((ext_vector_type(16))) _Float16 v16h;
typedef __attribute__((ext_vector_type(8)))  _Float16 v8h;
typedef __attribute__((ext_vector_type(4)))  _Float16 v4h;
typedef __attribute__((ext_vector_type(8)))  float    v8f;
typedef __attribute__((ext_vector_type(4)))  float    v4f;
typedef __attribute__((ext_vector_type(4)))  int      v4i;

#if __has_builtin(__builtin_amdgcn_global_load_async_to_lds_b128) && \
    __has_builtin(__builtin_amdgcn_s_wait_asynccnt)
#define USE_ASYNC_LDS 1
typedef __attribute__((address_space(1))) v4i gas_v4i;
typedef __attribute__((address_space(3))) v4i las_v4i;
#else
#define USE_ASYNC_LDS 0
#endif

// Raw 16-byte global->LDS copy; async (ASYNCcnt) when available.
static __device__ inline void cp_b128_to_lds(const _Float16* g, _Float16* l) {
#if USE_ASYNC_LDS
  // Global generic and AS(1) addresses are numerically identical -> via uintptr_t
  // (also drops const). LDS needs a real addrspacecast to strip the aperture.
  __builtin_amdgcn_global_load_async_to_lds_b128(
      (gas_v4i*)(uintptr_t)g, (las_v4i*)l, 0, 0);
#else
  *(v8h*)l = *(const v8h*)g;
#endif
}
static __device__ inline void cp_lds_wait() {
#if USE_ASYNC_LDS
  __builtin_amdgcn_s_wait_asynccnt(0);
#endif
}

static __device__ inline v16h cat16(v8h lo, v8h hi) {
  return __builtin_shufflevector(lo, hi, 0,1,2,3,4,5,6,7,8,9,10,11,12,13,14,15);
}
static __device__ inline v8f vzero8() {
  v8f v;
#pragma unroll
  for (int i = 0; i < 8; ++i) v[i] = 0.0f;
  return v;
}
static __device__ inline v4h ld4h(const float* p) {
  v4f f = *(const v4f*)p;
  v4h h;
#pragma unroll
  for (int i = 0; i < 4; ++i) h[i] = (_Float16)f[i];
  return h;
}
static __device__ inline v4h ld4h(const _Float16* p) { return *(const v4h*)p; }

// ---------------------------------------------------------------------------
// Tiled GEMM block:  Y[M,N] = A[M,K] * W[N,K]^T (+ bias), K=N=1024.
// Block tile 128x128, 8 waves (4 M-waves x 2 N-waves), wave tile 32x64.
// LDS tiles padded to 56 halves/row: 112B row stride => 16B-aligned b128 ops,
// and 16 consecutive rows map to distinct bank groups (stride 28 banks).
// ---------------------------------------------------------------------------
template <typename AT, bool BIAS, typename OT>
static __device__ inline void gemm_block(const AT* __restrict__ A,
                                         const float* __restrict__ W,
                                         const float* __restrict__ bias,
                                         OT* __restrict__ Y) {
  constexpr int BK  = 32;
  constexpr int LDT = 56;   // padded row stride in halves
  constexpr int K   = 1024;
  constexpr int N   = 1024;

  __shared__ __align__(16) _Float16 sA[128 * LDT];
  __shared__ __align__(16) _Float16 sB[128 * LDT];

  const int tid  = threadIdx.x;        // 0..255
  const int lane = tid & 31;
  const int wave = tid >> 5;           // 0..7
  const int wm   = wave & 3;           // 0..3 : 32-row strip
  const int wn   = wave >> 2;          // 0..1 : 64-col strip
  const int lh   = lane >> 4;          // half-wave (0/1)
  const int l16  = lane & 15;
  const int m0   = blockIdx.y * 128;
  const int n0   = blockIdx.x * 128;

  const int lrow = tid >> 3;           // 0..31
  const int lcol = (tid & 7) * 4;      // 0,4,..,28

  v8f acc[2][4];
#pragma unroll
  for (int i = 0; i < 2; ++i)
#pragma unroll
    for (int j = 0; j < 4; ++j) acc[i][j] = vzero8();

  for (int k0 = 0; k0 < K; k0 += BK) {
    __syncthreads();
    // Stage A (f32 or f16 -> f16) and W (f32 -> f16), 128x32 each.
#pragma unroll
    for (int p = 0; p < 4; ++p) {
      const int r = lrow + p * 32;
      *(v4h*)&sA[r * LDT + lcol] = ld4h(A + (size_t)(m0 + r) * K + k0 + lcol);
      *(v4h*)&sB[r * LDT + lcol] = ld4h(W + (size_t)(n0 + r) * K + k0 + lcol);
    }
    __syncthreads();

    // A fragments: lane holds row M=l16; halves 0..7 -> K=lh*8+0..7,
    // halves 8..15 -> K=16+lh*8+0..7 (ISA 16-bit A 16x32 layout).
    v16h af[2];
#pragma unroll
    for (int i = 0; i < 2; ++i) {
      const _Float16* base = &sA[(wm * 32 + i * 16 + l16) * LDT + lh * 8];
      af[i] = cat16(*(const v8h*)base, *(const v8h*)(base + 16));
    }
    // B fragments: lane holds col N=l16; halves 0..7 -> K=lh*8+0..7,
    // halves 8..15 -> K=16+lh*8+0..7 (same order as A).
    v16h bf[4];
#pragma unroll
    for (int j = 0; j < 4; ++j) {
      const _Float16* base = &sB[(wn * 64 + j * 16 + l16) * LDT + lh * 8];
      bf[j] = cat16(*(const v8h*)base, *(const v8h*)(base + 16));
    }
#pragma unroll
    for (int i = 0; i < 2; ++i)
#pragma unroll
      for (int j = 0; j < 4; ++j)
        acc[i][j] = __builtin_amdgcn_wmma_f32_16x16x32_f16(
            false, af[i], false, bf[j], (short)0, acc[i][j], false, false);
  }

  // Epilogue: C layout -> lane holds col n=l16, rows r + 8*lh across VGPRs.
#pragma unroll
  for (int i = 0; i < 2; ++i)
#pragma unroll
    for (int j = 0; j < 4; ++j) {
      const int n = n0 + wn * 64 + j * 16 + l16;
      float bv = 0.0f;
      if constexpr (BIAS) bv = bias[n];
#pragma unroll
      for (int r = 0; r < 8; ++r) {
        const int m = m0 + wm * 32 + i * 16 + r + 8 * lh;
        Y[(size_t)m * N + n] = (OT)(acc[i][j][r] + bv);
      }
    }
}

__global__ __launch_bounds__(256) void qkv_proj_kernel(
    const float* __restrict__ x, const float* __restrict__ Wq,
    const float* __restrict__ Wk, const float* __restrict__ Wv,
    _Float16* __restrict__ Q, _Float16* __restrict__ K,
    _Float16* __restrict__ V) {
  const float* W = (blockIdx.z == 0) ? Wq : (blockIdx.z == 1) ? Wk : Wv;
  _Float16*    Y = (blockIdx.z == 0) ? Q  : (blockIdx.z == 1) ? K  : V;
  gemm_block<float, false, _Float16>(x, W, nullptr, Y);
}

__global__ __launch_bounds__(256) void out_proj_kernel(
    const _Float16* __restrict__ ctx, const float* __restrict__ Wo,
    const float* __restrict__ bo, float* __restrict__ out) {
  gemm_block<_Float16, true, float>(ctx, Wo, bo, out);
}

// ---------------------------------------------------------------------------
// Flash attention: one block per (64 q-rows, head, batch); 4 waves x 16 rows.
// LDS rows padded to 88 halves (176B): 16B-aligned b128, conflict-free frags.
// Q/K tiles staged with async global->LDS copies (no VGPR round trip).
// ---------------------------------------------------------------------------
__global__ __launch_bounds__(128) void attn_kernel(
    const _Float16* __restrict__ Qg, const _Float16* __restrict__ Kg,
    const _Float16* __restrict__ Vg, _Float16* __restrict__ Cg) {
  constexpr int LD = 88;
  __shared__ __align__(16) _Float16 sQ[64 * LD];
  __shared__ __align__(16) _Float16 sK[64 * LD];
  __shared__ __align__(16) _Float16 sV[64 * LD];        // V^T: [hd][key]
  __shared__ __align__(16) _Float16 sP[4 * 16 * LD];    // per-wave P tiles

  const int tid  = threadIdx.x;   // 0..127
  const int lane = tid & 31;
  const int wave = tid >> 5;      // 0..3
  const int lh   = lane >> 4;
  const int l16  = lane & 15;
  const int qb   = blockIdx.x;    // 0..31
  const int h    = blockIdx.y;
  const int b    = blockIdx.z;
  const size_t headOff = (size_t)b * T_ * D_ + (size_t)h * HD_;

  // Async-stage the 64x64 Q tile (raw f16 bytes).
#pragma unroll
  for (int p = 0; p < 4; ++p) {
    const int c = tid + p * 128, row = c >> 3, ch = c & 7;
    cp_b128_to_lds(Qg + headOff + (size_t)(qb * 64 + row) * D_ + ch * 8,
                   &sQ[row * LD + ch * 8]);
  }
  cp_lds_wait();
  __syncthreads();

  // Per-wave Q fragments; fold the 1/sqrt(HD)=0.125 scale into the registers.
  v16h aq[2];
#pragma unroll
  for (int kk = 0; kk < 2; ++kk) {
    const _Float16* base = &sQ[(wave * 16 + l16) * LD + kk * 32 + lh * 8];
    v16h t = cat16(*(const v8h*)base, *(const v8h*)(base + 16));
#pragma unroll
    for (int e = 0; e < 16; ++e) t[e] = t[e] * (_Float16)0.125f;
    aq[kk] = t;
  }

  float rmax[8], rsum[8];
#pragma unroll
  for (int r = 0; r < 8; ++r) { rmax[r] = -3.0e38f; rsum[r] = 0.0f; }
  v8f o[4];
#pragma unroll
  for (int j = 0; j < 4; ++j) o[j] = vzero8();

  for (int kb = 0; kb < T_ / 64; ++kb) {
    __syncthreads();   // protect sK/sV/sP from previous iteration readers
#pragma unroll
    for (int p = 0; p < 4; ++p) {
      const int c = tid + p * 128, row = c >> 3, ch = c & 7;
      // K tile: raw async copy into LDS.
      cp_b128_to_lds(Kg + headOff + (size_t)(kb * 64 + row) * D_ + ch * 8,
                     &sK[row * LD + ch * 8]);
      // V tile: through VGPRs with transpose scatter ([key][hd]->[hd][key]).
      v8h v = *(const v8h*)(Vg + headOff + (size_t)(kb * 64 + row) * D_ + ch * 8);
#pragma unroll
      for (int e = 0; e < 8; ++e) sV[(ch * 8 + e) * LD + row] = v[e];
    }
    // Prefetch next K/V block rows into cache while we compute on this one.
    if (kb + 1 < T_ / 64) {
      const size_t nrow = headOff + (size_t)((kb + 1) * 64 + (tid & 63)) * D_;
      __builtin_prefetch(Kg + nrow, 0, 3);
      __builtin_prefetch(Vg + nrow, 0, 3);
    }
    cp_lds_wait();
    __syncthreads();

    // S = Q * K^T  (GEMM-K = hd = 64 => 2 wmma k-steps per n-tile)
    v8f s[4];
#pragma unroll
    for (int j = 0; j < 4; ++j) {
      s[j] = vzero8();
#pragma unroll
      for (int kk = 0; kk < 2; ++kk) {
        const _Float16* base = &sK[(j * 16 + l16) * LD + kk * 32 + lh * 8];
        v16h bk = cat16(*(const v8h*)base, *(const v8h*)(base + 16));
        s[j] = __builtin_amdgcn_wmma_f32_16x16x32_f16(
            false, aq[kk], false, bk, (short)0, s[j], false, false);
      }
    }

    // Online softmax: row m = r + 8*lh lives entirely in this 16-lane half.
    float fac[8];
#pragma unroll
    for (int r = 0; r < 8; ++r) {
      float mx = fmaxf(fmaxf(s[0][r], s[1][r]), fmaxf(s[2][r], s[3][r]));
#pragma unroll
      for (int off = 1; off < 16; off <<= 1)
        mx = fmaxf(mx, __shfl_xor(mx, off, 16));
      const float nm = fmaxf(rmax[r], mx);
      fac[r] = __expf(rmax[r] - nm);
      rmax[r] = nm;
      float ls = 0.0f;
#pragma unroll
      for (int j = 0; j < 4; ++j) {
        const float pv = __expf(s[j][r] - nm);
        s[j][r] = pv;
        ls += pv;
      }
#pragma unroll
      for (int off = 1; off < 16; off <<= 1)
        ls += __shfl_xor(ls, off, 16);
      rsum[r] = rsum[r] * fac[r] + ls;
#pragma unroll
      for (int j = 0; j < 4; ++j) o[j][r] *= fac[r];
    }

    // Re-layout P (C-layout) -> A-operand via per-wave LDS tile.
#pragma unroll
    for (int j = 0; j < 4; ++j)
#pragma unroll
      for (int r = 0; r < 8; ++r)
        sP[(wave * 16 + r + 8 * lh) * LD + j * 16 + l16] = (_Float16)s[j][r];
    __syncthreads();

    // O += P * V   (GEMM-K = 64 keys => 2 wmma k-steps)
#pragma unroll
    for (int kk = 0; kk < 2; ++kk) {
      const _Float16* pb = &sP[(wave * 16 + l16) * LD + kk * 32 + lh * 8];
      v16h ap = cat16(*(const v8h*)pb, *(const v8h*)(pb + 16));
#pragma unroll
      for (int j = 0; j < 4; ++j) {
        const _Float16* vb = &sV[(j * 16 + l16) * LD + kk * 32 + lh * 8];
        v16h bv = cat16(*(const v8h*)vb, *(const v8h*)(vb + 16));
        o[j] = __builtin_amdgcn_wmma_f32_16x16x32_f16(
            false, ap, false, bv, (short)0, o[j], false, false);
      }
    }
  }

  // Normalize and store ctx (f16, [b,t,h,hd] interleaved layout).
#pragma unroll
  for (int j = 0; j < 4; ++j)
#pragma unroll
    for (int r = 0; r < 8; ++r) {
      const int t  = qb * 64 + wave * 16 + r + 8 * lh;
      const int hd = j * 16 + l16;
      Cg[headOff + (size_t)t * D_ + hd] = (_Float16)(o[j][r] / rsum[r]);
    }
}

// ---------------------------------------------------------------------------
extern "C" void kernel_launch(void* const* d_in, const int* in_sizes, int n_in,
                              void* d_out, int out_size, void* d_ws, size_t ws_size,
                              hipStream_t stream) {
  const float* x  = (const float*)d_in[0];
  const float* Wq = (const float*)d_in[1];
  const float* Wk = (const float*)d_in[2];
  const float* Wv = (const float*)d_in[3];
  const float* Wo = (const float*)d_in[4];
  const float* bo = (const float*)d_in[5];
  float* out = (float*)d_out;

  const size_t tbytes = (size_t)M_ * D_ * sizeof(_Float16);  // 8 MB
  char* wsb = (char*)d_ws;
  _Float16* Qh = (_Float16*)(wsb + 0 * tbytes);
  _Float16* Kh = (_Float16*)(wsb + 1 * tbytes);
  _Float16* Vh = (_Float16*)(wsb + 2 * tbytes);
  _Float16* Ch = (_Float16*)(wsb + 3 * tbytes);

  dim3 gQKV(D_ / 128, M_ / 128, 3);      // (8, 32, 3)
  qkv_proj_kernel<<<gQKV, 256, 0, stream>>>(x, Wq, Wk, Wv, Qh, Kh, Vh);

  dim3 gAttn(T_ / 64, H_, B_);           // (32, 16, 2)
  attn_kernel<<<gAttn, 128, 0, stream>>>(Qh, Kh, Vh, Ch);

  dim3 gOut(D_ / 128, M_ / 128, 1);      // (8, 32)
  out_proj_kernel<<<gOut, 256, 0, stream>>>(Ch, Wo, bo, out);
}
